// GTrXL_63591285784723
// MI455X (gfx1250) — compile-verified, matrix operand order checked by reading, not run on hardware
//
#include <hip/hip_runtime.h>
#include <hip/hip_bf16.h>
#include <math.h>

// Problem constants (GTrXL reference)
#define CUR   512
#define MEMN  256
#define FULLN 768
#define BSZ   32
#define DMODEL 512
#define NH    8
#define DH    64
#define INDIM 256
#define NL    4

typedef unsigned short u16;
typedef unsigned int   u32;

typedef __attribute__((ext_vector_type(16))) __bf16 bf16x16;
typedef __attribute__((ext_vector_type(8)))  float  f32x8;

// ---- gfx1250 async global->LDS staging support -----------------------------
#if defined(__AMDGCN__) && __has_builtin(__builtin_amdgcn_global_load_async_to_lds_b128)
#define ASYNC_MODE 1      // clang builtin
#elif defined(__AMDGCN__) && defined(__gfx1250__)
#define ASYNC_MODE 2      // inline asm fallback
#else
#define ASYNC_MODE 0      // plain register staging
#endif

#if defined(__AMDGCN__)
typedef int v4i_gcc __attribute__((vector_size(16)));
typedef __attribute__((address_space(1))) v4i_gcc* gv4i_p;
typedef __attribute__((address_space(3))) v4i_gcc* lv4i_p;
#endif

__device__ __forceinline__ void async_copy16(const u16* gp, u16* lp) {
#if ASYNC_MODE == 1
    __builtin_amdgcn_global_load_async_to_lds_b128(
        (gv4i_p)(void*)(const_cast<u16*>(gp)),
        (lv4i_p)(void*)lp, 0, 0);
#elif ASYNC_MODE == 2
    u32 laddr = (u32)(unsigned long long)(lv4i_p)(void*)lp;
    unsigned long long gaddr = (unsigned long long)gp;
    asm volatile("global_load_async_to_lds_b128 %0, %1, off"
                 :: "v"(laddr), "v"(gaddr) : "memory");
#else
    (void)gp; (void)lp;
#endif
}

__device__ __forceinline__ void wait_async_lds() {
#if ASYNC_MODE
#if __has_builtin(__builtin_amdgcn_s_wait_asynccnt)
    __builtin_amdgcn_s_wait_asynccnt(0);
#else
    asm volatile("s_wait_asynccnt 0" ::: "memory");
#endif
#endif
}

__device__ __forceinline__ u16 f2bf(float f) {
    u32 u = __float_as_uint(f);
    u32 r = u + 0x7FFFu + ((u >> 16) & 1u);   // round-to-nearest-even
    return (u16)(r >> 16);
}

// ---------------------------------------------------------------------------
// fp32 -> bf16 conversion (grid-stride)
// ---------------------------------------------------------------------------
__global__ void k_cvt(const float* __restrict__ s, u16* __restrict__ d, long n) {
    long i = (long)blockIdx.x * blockDim.x + threadIdx.x;
    long st = (long)gridDim.x * blockDim.x;
    for (; i < n; i += st) d[i] = f2bf(s[i]);
}

// ---------------------------------------------------------------------------
// positional encoding pe(FULLN, D) in bf16
// ---------------------------------------------------------------------------
__global__ void k_pe(u16* __restrict__ pe) {
    int idx = blockIdx.x * blockDim.x + threadIdx.x;
    if (idx >= FULLN * DMODEL) return;
    int p = idx / DMODEL, c = idx % DMODEL;
    float pos = (float)(FULLN - 1 - p);
    int k = (c < DMODEL / 2) ? c : (c - DMODEL / 2);
    float invf = __powf(10000.0f, -2.0f * (float)k / (float)DMODEL);
    float s = pos * invf;
    pe[idx] = f2bf((c < DMODEL / 2) ? __sinf(s) : __cosf(s));
}

// ---------------------------------------------------------------------------
// LayerNorm over D=512; rows [0,rowsA) from srcA, rest from srcB; bf16 out
// ---------------------------------------------------------------------------
__global__ void k_ln(const float* __restrict__ srcA, int rowsA,
                     const float* __restrict__ srcB,
                     const float* __restrict__ g, const float* __restrict__ b,
                     u16* __restrict__ out) {
    int row = blockIdx.x;
    const float* src = (row < rowsA) ? (srcA + (long)row * DMODEL)
                                     : (srcB + (long)(row - rowsA) * DMODEL);
    int tid = threadIdx.x;
    float x0 = src[tid], x1 = src[tid + 256];
    __shared__ float red[256];
    red[tid] = x0 + x1;
    __syncthreads();
    for (int s = 128; s > 0; s >>= 1) { if (tid < s) red[tid] += red[tid + s]; __syncthreads(); }
    float mean = red[0] * (1.0f / DMODEL);
    __syncthreads();
    float d0 = x0 - mean, d1 = x1 - mean;
    red[tid] = d0 * d0 + d1 * d1;
    __syncthreads();
    for (int s = 128; s > 0; s >>= 1) { if (tid < s) red[tid] += red[tid + s]; __syncthreads(); }
    float rstd = rsqrtf(red[0] * (1.0f / DMODEL) + 1e-5f);
    long o = (long)row * DMODEL;
    out[o + tid]       = f2bf(d0 * rstd * g[tid]       + b[tid]);
    out[o + tid + 256] = f2bf(d1 * rstd * g[tid + 256] + b[tid + 256]);
}

// ---------------------------------------------------------------------------
// pack q(+u)/(+v) into per-(b,h) row-major (CUR x DH) bf16 tiles
// ---------------------------------------------------------------------------
__global__ void k_pack_qu(const float* __restrict__ q, const float* __restrict__ u,
                          const float* __restrict__ v, u16* __restrict__ Qu,
                          u16* __restrict__ Qv) {
    long n = (long)CUR * BSZ * NH * DH;
    long i = (long)blockIdx.x * blockDim.x + threadIdx.x;
    long st = (long)gridDim.x * blockDim.x;
    for (; i < n; i += st) {
        int d  = (int)(i & (DH - 1));
        long t = i >> 6;
        int ii = (int)(t & (CUR - 1));
        t >>= 9;
        int h = (int)(t & (NH - 1));
        int b = (int)(t >> 3);
        float qq = q[((long)ii * BSZ + b) * (NH * DH) + h * DH + d];
        Qu[i] = f2bf(qq + u[h * DH + d]);
        Qv[i] = f2bf(qq + v[h * DH + d]);
    }
}

// ---------------------------------------------------------------------------
// pack kv(bf16, rows j*BSZ+b, cols 2*H*DH) -> Kt[b][h][d][j], Val[b][h][j][d]
// ---------------------------------------------------------------------------
__global__ void k_pack_kv(const u16* __restrict__ kv, u16* __restrict__ Kt,
                          u16* __restrict__ Val) {
    long n = (long)FULLN * BSZ * NH * DH;
    long i = (long)blockIdx.x * blockDim.x + threadIdx.x;
    long st = (long)gridDim.x * blockDim.x;
    for (; i < n; i += st) {
        // Kt linear:  i = ((b*NH+h)*DH + d)*FULLN + j
        int j  = (int)(i % FULLN);
        long r1 = i / FULLN;
        int d  = (int)(r1 & (DH - 1));
        long r2 = r1 >> 6;
        int h  = (int)(r2 & (NH - 1));
        int b  = (int)(r2 >> 3);
        Kt[i] = kv[((long)j * BSZ + b) * (2 * NH * DH) + h * DH + d];
        // Val linear: i = ((b*NH+h)*FULLN + j)*DH + d
        int d2 = (int)(i & (DH - 1));
        long s1 = i >> 6;
        int j2 = (int)(s1 % FULLN);
        long s2 = s1 / FULLN;
        int h2 = (int)(s2 & (NH - 1));
        int b2 = (int)(s2 >> 3);
        Val[i] = kv[((long)j2 * BSZ + b2) * (2 * NH * DH) + NH * DH + h2 * DH + d2];
    }
}

// ---------------------------------------------------------------------------
// pack r(FULLN, H*DH) f32 -> Rt[h][d][j] bf16
// ---------------------------------------------------------------------------
__global__ void k_pack_r(const float* __restrict__ r, u16* __restrict__ Rt) {
    long n = (long)NH * DH * FULLN;
    long i = (long)blockIdx.x * blockDim.x + threadIdx.x;
    long st = (long)gridDim.x * blockDim.x;
    for (; i < n; i += st) {
        int j  = (int)(i % FULLN);
        int hd = (int)(i / FULLN);
        Rt[i] = f2bf(r[(long)j * (NH * DH) + hd]);
    }
}

// ---------------------------------------------------------------------------
// masked softmax with on-the-fly rel_shift gather; probs bf16 out
// grid(CUR, NH), block 256. cont/posr: [h][i][j] chunks for one batch b.
// ---------------------------------------------------------------------------
__global__ void k_softmax(const float* __restrict__ cont, const float* __restrict__ posr,
                          u16* __restrict__ probs) {
    int i = blockIdx.x, h = blockIdx.y, tid = threadIdx.x;
    const float* crow  = cont  + ((long)h * CUR + i) * FULLN;
    const float* pbase = posr  + (long)h * CUR * FULLN;
    u16*         orow  = probs + ((long)h * CUR + i) * FULLN;
    float sv[3];
    bool  va[3];
    float mx = -3.4e38f;
    #pragma unroll
    for (int t = 0; t < 3; ++t) {
        int j = tid + t * 256;
        bool valid = (j <= i + MEMN);
        float s = -3.4e38f;
        if (valid) {
            long flat = (long)CUR + (long)i * FULLN + j;
            int a  = (int)(flat / (FULLN + 1));
            int bb = (int)(flat % (FULLN + 1));
            float p = (bb > 0) ? pbase[(long)a * FULLN + (bb - 1)] : 0.0f;
            s = (crow[j] + p) * 0.125f;   // 1/sqrt(DH)
        }
        sv[t] = s; va[t] = valid;
        mx = fmaxf(mx, s);
    }
    __shared__ float red[256];
    red[tid] = mx; __syncthreads();
    for (int s = 128; s > 0; s >>= 1) { if (tid < s) red[tid] = fmaxf(red[tid], red[tid + s]); __syncthreads(); }
    mx = red[0]; __syncthreads();
    float ev[3], sum = 0.f;
    #pragma unroll
    for (int t = 0; t < 3; ++t) { ev[t] = va[t] ? __expf(sv[t] - mx) : 0.f; sum += ev[t]; }
    red[tid] = sum; __syncthreads();
    for (int s = 128; s > 0; s >>= 1) { if (tid < s) red[tid] += red[tid + s]; __syncthreads(); }
    float inv = 1.0f / red[0];
    #pragma unroll
    for (int t = 0; t < 3; ++t) orow[tid + t * 256] = f2bf(ev[t] * inv);
}

// ---------------------------------------------------------------------------
// elementwise: rx = bf16(r * x)   (r already sigmoid'ed by GEMM epilogue)
// ---------------------------------------------------------------------------
__global__ void k_ew_rx(const float* __restrict__ r, const float* __restrict__ x,
                        u16* __restrict__ out, long n) {
    long i = (long)blockIdx.x * blockDim.x + threadIdx.x;
    long st = (long)gridDim.x * blockDim.x;
    for (; i < n; i += st) out[i] = f2bf(r[i] * x[i]);
}

// elementwise: o = (1-z)*x + z*g ; write f32 + bf16 (z sigmoid'ed, g tanh'ed)
__global__ void k_ew_gateout(const float* __restrict__ z, const float* __restrict__ g,
                             const float* __restrict__ x, float* __restrict__ of,
                             u16* __restrict__ ob, long n) {
    long i = (long)blockIdx.x * blockDim.x + threadIdx.x;
    long st = (long)gridDim.x * blockDim.x;
    for (; i < n; i += st) {
        float zz = z[i];
        float o = (1.0f - zz) * x[i] + zz * g[i];
        of[i] = o;
        ob[i] = f2bf(o);
    }
}

// ---------------------------------------------------------------------------
// Batched bf16 WMMA GEMM: C[batch] = act(A[batch] @ B[batch] (+C) (+bias*s))
// A: (M,K) bf16 row-major, lda. B: (K,N) bf16 row-major, ldb.
// C f32 (optional), Cb bf16 mirror (optional), shared ldc/strideC.
// Block tile 128x128, 8 wave32 waves (wave tile 64x32 = 4x2 WMMA 16x16x32).
// A tile staged via GLOBAL_LOAD_ASYNC_TO_LDS_B128 (ASYNCcnt pipeline).
// ---------------------------------------------------------------------------
#define BM 128
#define BN 128
#define BK 32
#define SA 40   // padded LDS row stride in shorts (80B, 16B aligned)

__global__ __launch_bounds__(256) void k_gemm(
    const u16* __restrict__ A, long lda, long sA,
    const u16* __restrict__ B, long ldb, long sB,
    float* __restrict__ C, u16* __restrict__ Cb, long ldc, long sC,
    const float* __restrict__ bias, float bscale,
    int M, int N, int K, int beta1, int act)
{
    __shared__ u16 smA[2][BM * SA];
    __shared__ u16 smB[2][BN * SA];

    int tid  = threadIdx.x;
    int lane = tid & 31, wid = tid >> 5;
    int wm = (wid & 1) * 64;
    int wn = (wid >> 1) * 32;
    int bm0 = blockIdx.x * BM, bn0 = blockIdx.y * BN;
    int batch = blockIdx.z;
    const u16* Ab = A + (long)batch * sA;
    const u16* Bb = B + (long)batch * sB;
    int nk = K / BK;

    f32x8 acc[4][2];
    #pragma unroll
    for (int a = 0; a < 4; ++a)
        #pragma unroll
        for (int b = 0; b < 2; ++b)
            #pragma unroll
            for (int r = 0; r < 8; ++r) acc[a][b][r] = 0.f;

    union Frag { u32 u[8]; bf16x16 v; };
    union BLd  { uint4 q; u16 s[8]; };

    auto stage = [&](int buf, int kt) {
        int k0 = kt * BK;
        // ---- A tile: BMx32 row-major; 512 chunks of 16B, 2 per thread ----
        #pragma unroll
        for (int t = 0; t < 2; ++t) {
            int c = tid + t * 256;           // 0..511
            int row = c >> 2;                // 4 chunks per row
            int kq  = (c & 3) << 3;          // element offset, *8
            int grow = bm0 + row;
            u16* lp = &smA[buf][row * SA + kq];
#if ASYNC_MODE
            if (grow < M) {
                async_copy16(Ab + (long)grow * lda + k0 + kq, lp);
            } else {
                *(uint4*)lp = make_uint4(0u, 0u, 0u, 0u);
            }
#else
            BLd av; av.q = make_uint4(0u, 0u, 0u, 0u);
            if (grow < M) av.q = *(const uint4*)(Ab + (long)grow * lda + k0 + kq);
            *(uint4*)lp = av.q;
#endif
        }
        // speculative prefetch of the tile after next (global_prefetch_b8)
        if (kt + 2 < nk) {
            int grow = bm0 + (tid >> 1);
            if (grow < M)
                __builtin_prefetch((const void*)(Ab + (long)grow * lda + (kt + 2) * BK), 0, 3);
        }
        // ---- B tile: load (32 x BN) rows, store transposed Bt[n][k] ----
        #pragma unroll
        for (int t = 0; t < 2; ++t) {
            int c = tid + t * 256;           // 0..511
            int krow = c >> 4;               // 0..31
            int nq   = (c & 15) << 3;        // *8
            const u16* p = Bb + (long)(k0 + krow) * ldb + bn0 + nq;
            BLd tv;
            if (bn0 + nq + 7 < N) {
                tv.q = *(const uint4*)p;
            } else {
                #pragma unroll
                for (int e = 0; e < 8; ++e)
                    tv.s[e] = (bn0 + nq + e < N) ? p[e] : (u16)0;
            }
            #pragma unroll
            for (int e = 0; e < 8; ++e)
                smB[buf][(nq + e) * SA + krow] = tv.s[e];
        }
    };

    int lm = lane & 15, hf = lane >> 4;

    stage(0, 0);
    wait_async_lds();
    __syncthreads();
    for (int kt = 0; kt < nk; ++kt) {
        int buf = kt & 1;
        if (kt + 1 < nk) stage(buf ^ 1, kt + 1);

        Frag fa[4], fb[2];
        #pragma unroll
        for (int tm = 0; tm < 4; ++tm) {
            const u16* base = &smA[buf][(wm + tm * 16 + lm) * SA + hf * 8];
            #pragma unroll
            for (int i2 = 0; i2 < 4; ++i2) {
                fa[tm].u[i2]     = *(const u32*)(base + 2 * i2);       // K = hf*8 + 2*i2
                fa[tm].u[4 + i2] = *(const u32*)(base + 16 + 2 * i2);  // K = 16 + hf*8 + 2*i2
            }
        }
        #pragma unroll
        for (int tn = 0; tn < 2; ++tn) {
            const u16* base = &smB[buf][(wn + tn * 16 + lm) * SA + hf * 8];
            #pragma unroll
            for (int i2 = 0; i2 < 4; ++i2) {
                fb[tn].u[i2]     = *(const u32*)(base + 2 * i2);
                fb[tn].u[4 + i2] = *(const u32*)(base + 16 + 2 * i2);
            }
        }
        #pragma unroll
        for (int tm = 0; tm < 4; ++tm)
            #pragma unroll
            for (int tn = 0; tn < 2; ++tn)
                acc[tm][tn] = __builtin_amdgcn_wmma_f32_16x16x32_bf16(
                    false, fa[tm].v, false, fb[tn].v,
                    (short)0, acc[tm][tn], false, false);
        wait_async_lds();
        __syncthreads();
    }

    // ---- epilogue: C layout lanes 0-15 N=lane, VGPR r -> M = r + 8*(lane>=16) ----
    #pragma unroll
    for (int tm = 0; tm < 4; ++tm) {
        #pragma unroll
        for (int tn = 0; tn < 2; ++tn) {
            int col = bn0 + wn + tn * 16 + lm;
            if (col >= N) continue;
            #pragma unroll
            for (int r = 0; r < 8; ++r) {
                int row = bm0 + wm + tm * 16 + hf * 8 + r;
                if (row >= M) continue;
                float v = acc[tm][tn][r];
                long off = (long)batch * sC + (long)row * ldc + col;
                if (beta1) v += C[off];
                if (bias)  v += bscale * bias[col];
                if      (act == 1) v = fmaxf(v, 0.f);
                else if (act == 2) v = 1.0f / (1.0f + __expf(-v));
                else if (act == 3) v = tanhf(v);
                if (C)  C[off]  = v;
                if (Cb) Cb[off] = f2bf(v);
            }
        }
    }
}

// ---------------------------------------------------------------------------
// host orchestration
// ---------------------------------------------------------------------------
extern "C" void kernel_launch(void* const* d_in, const int* in_sizes, int n_in,
                              void* d_out, int out_size, void* d_ws, size_t ws_size,
                              hipStream_t stream) {
    (void)in_sizes; (void)n_in; (void)out_size; (void)ws_size;
    const float* x_in   = (const float*)d_in[0];
    const float* memory = (const float*)d_in[1];
    const float* W_emb  = (const float*)d_in[2];
    const float* b_emb  = (const float*)d_in[3];
    const float* u_in   = (const float*)d_in[4];
    const float* v_in   = (const float*)d_in[5];
    const float* ln1_g  = (const float*)d_in[6];
    const float* ln1_b  = (const float*)d_in[7];
    const float* ln2_g  = (const float*)d_in[8];
    const float* ln2_b  = (const float*)d_in[9];
    const float* W_kv   = (const float*)d_in[10];
    const float* b_kv   = (const float*)d_in[11];
    const float* W_q    = (const float*)d_in[12];
    const float* b_q    = (const float*)d_in[13];
    const float* W_proj = (const float*)d_in[14];
    const float* b_proj = (const float*)d_in[15];
    const float* W_pos  = (const float*)d_in[16];
    const float* b_pos  = (const float*)d_in[17];
    const float* g1_w   = (const float*)d_in[18];
    const float* g1_b   = (const float*)d_in[19];
    const float* g2_w   = (const float*)d_in[20];
    const float* g2_b   = (const float*)d_in[21];
    const float* W_m1   = (const float*)d_in[22];
    const float* b_m1   = (const float*)d_in[23];
    const float* W_m2   = (const float*)d_in[24];
    const float* b_m2   = (const float*)d_in[25];

    const long Mc = (long)CUR * BSZ;    // 16384
    const long Mf = (long)FULLN * BSZ;  // 24576
    const long D  = DMODEL;
    const long HD = (long)NH * DH;      // 512

    char* base = (char*)d_ws;
    size_t off = 0;
    auto alloc = [&](size_t bytes) -> void* {
        off = (off + 255) & ~(size_t)255;
        void* p = base + off;
        off += bytes;
        return p;
    };

    // bf16 weight copies
    u16* Wemb_bf  = (u16*)alloc(sizeof(u16) * INDIM * D);
    u16* Wkv_bf   = (u16*)alloc(sizeof(u16) * NL * D * 2 * HD);
    u16* Wq_bf    = (u16*)alloc(sizeof(u16) * NL * D * HD);
    u16* Wpos_bf  = (u16*)alloc(sizeof(u16) * NL * D * HD);
    u16* Wproj_bf = (u16*)alloc(sizeof(u16) * NL * HD * D);
    u16* G1_bf    = (u16*)alloc(sizeof(u16) * NL * 6 * D * D);
    u16* G2_bf    = (u16*)alloc(sizeof(u16) * NL * 6 * D * D);
    u16* Wm1_bf   = (u16*)alloc(sizeof(u16) * NL * D * D);
    u16* Wm2_bf   = (u16*)alloc(sizeof(u16) * NL * D * D);
    u16* x_bf     = (u16*)alloc(sizeof(u16) * Mc * INDIM);
    u16* pe_bf    = (u16*)alloc(sizeof(u16) * FULLN * D);

    // activations / scratch
    float* h_f    = (float*)alloc(sizeof(float) * Mc * D);
    u16*   h_bf   = (u16*)  alloc(sizeof(u16)   * Mc * D);
    u16*   full_bf= (u16*)  alloc(sizeof(u16)   * Mf * D);
    u16*   kv_bf  = (u16*)  alloc(sizeof(u16)   * Mf * 2 * HD);
    float* q_f    = (float*)alloc(sizeof(float) * Mc * HD);
    u16*   Qu     = (u16*)  alloc(sizeof(u16)   * (long)BSZ * NH * CUR * DH);
    u16*   Qv     = (u16*)  alloc(sizeof(u16)   * (long)BSZ * NH * CUR * DH);
    u16*   Kt     = (u16*)  alloc(sizeof(u16)   * (long)BSZ * NH * DH * FULLN);
    u16*   Val    = (u16*)  alloc(sizeof(u16)   * (long)BSZ * NH * FULLN * DH);
    float* r_f    = (float*)alloc(sizeof(float) * FULLN * HD);
    u16*   Rt     = (u16*)  alloc(sizeof(u16)   * (long)NH * DH * FULLN);
    float* cont_f = (float*)alloc(sizeof(float) * (long)NH * CUR * FULLN);
    float* posr_f = (float*)alloc(sizeof(float) * (long)NH * CUR * FULLN);
    u16*   probs  = (u16*)  alloc(sizeof(u16)   * (long)NH * CUR * FULLN);
    u16*   av_bf  = (u16*)  alloc(sizeof(u16)   * Mc * HD);
    u16*   a1_bf  = (u16*)  alloc(sizeof(u16)   * Mc * D);
    float* rbuf   = (float*)alloc(sizeof(float) * Mc * D);
    float* zbuf   = (float*)alloc(sizeof(float) * Mc * D);
    float* gbuf   = (float*)alloc(sizeof(float) * Mc * D);
    u16*   rx_bf  = (u16*)  alloc(sizeof(u16)   * Mc * D);
    float* o1_f   = (float*)alloc(sizeof(float) * Mc * D);
    u16*   o1_bf  = (u16*)  alloc(sizeof(u16)   * Mc * D);
    u16*   x2_bf  = (u16*)  alloc(sizeof(u16)   * Mc * D);
    u16*   m1_bf  = (u16*)  alloc(sizeof(u16)   * Mc * D);
    u16*   m_bf   = (u16*)  alloc(sizeof(u16)   * Mc * D);

    auto cvt = [&](const float* s, u16* d, long n) {
        int blocks = (int)((n + 1023) / 1024);
        if (blocks > 8192) blocks = 8192;
        if (blocks < 1) blocks = 1;
        k_cvt<<<dim3(blocks), dim3(256), 0, stream>>>(s, d, n);
    };

    auto gemm = [&](const u16* A, long lda, long sA,
                    const u16* B, long ldb, long sB,
                    float* C, u16* Cb, long ldc, long sC,
                    const float* bias, float bscale,
                    int M, int N, int K, int batch, int beta1, int act) {
        dim3 grid((M + BM - 1) / BM, (N + BN - 1) / BN, batch);
        k_gemm<<<grid, dim3(256), 0, stream>>>(A, lda, sA, B, ldb, sB,
                                               C, Cb, ldc, sC, bias, bscale,
                                               M, N, K, beta1, act);
    };

    // ---- weight/input conversion ----
    cvt(W_emb,  Wemb_bf,  (long)INDIM * D);
    cvt(W_kv,   Wkv_bf,   (long)NL * D * 2 * HD);
    cvt(W_q,    Wq_bf,    (long)NL * D * HD);
    cvt(W_pos,  Wpos_bf,  (long)NL * D * HD);
    cvt(W_proj, Wproj_bf, (long)NL * HD * D);
    cvt(g1_w,   G1_bf,    (long)NL * 6 * D * D);
    cvt(g2_w,   G2_bf,    (long)NL * 6 * D * D);
    cvt(W_m1,   Wm1_bf,   (long)NL * D * D);
    cvt(W_m2,   Wm2_bf,   (long)NL * D * D);
    cvt(x_in,   x_bf,     Mc * INDIM);
    k_pe<<<dim3((FULLN * DMODEL + 255) / 256), dim3(256), 0, stream>>>(pe_bf);

    // ---- embed: h = relu(x @ W_emb + b_emb) ----
    gemm(x_bf, INDIM, 0, Wemb_bf, D, 0, h_f, h_bf, D, 0,
         b_emb, 1.0f, (int)Mc, (int)D, INDIM, 1, 0, 1);

    const long nEl = Mc * D;
    const int ewBlocks = 8192;

    for (int l = 0; l < NL; ++l) {
        const u16* Wkv_l  = Wkv_bf  + (long)l * D * 2 * HD;
        const u16* Wq_l   = Wq_bf   + (long)l * D * HD;
        const u16* Wpos_l = Wpos_bf + (long)l * D * HD;
        const u16* Wpr_l  = Wproj_bf+ (long)l * HD * D;
        const u16* Wm1_l  = Wm1_bf  + (long)l * D * D;
        const u16* Wm2_l  = Wm2_bf  + (long)l * D * D;
        auto G1 = [&](int gate) { return G1_bf + ((long)l * 6 + gate) * D * D; };
        auto G2 = [&](int gate) { return G2_bf + ((long)l * 6 + gate) * D * D; };

        // x1 = LN(concat(memory[l], h))
        k_ln<<<dim3((unsigned)Mf), dim3(256), 0, stream>>>(
            memory + (long)l * MEMN * BSZ * D, MEMN * BSZ, h_f,
            ln1_g + l * D, ln1_b + l * D, full_bf);

        // kv = x1 @ W_kv + b_kv  (bf16 out)
        gemm(full_bf, D, 0, Wkv_l, 2 * HD, 0, nullptr, kv_bf, 2 * HD, 0,
             b_kv + (long)l * 2 * HD, 1.0f, (int)Mf, (int)(2 * HD), (int)D, 1, 0, 0);
        // q = h @ W_q + b_q  (f32 out)
        gemm(h_bf, D, 0, Wq_l, HD, 0, q_f, nullptr, HD, 0,
             b_q + l * HD, 1.0f, (int)Mc, (int)HD, (int)D, 1, 0, 0);
        // r = pe @ W_pos + b_pos
        gemm(pe_bf, D, 0, Wpos_l, HD, 0, r_f, nullptr, HD, 0,
             b_pos + l * HD, 1.0f, FULLN, (int)HD, (int)D, 1, 0, 0);

        k_pack_qu<<<dim3(8192), dim3(256), 0, stream>>>(q_f, u_in, v_in, Qu, Qv);
        k_pack_kv<<<dim3(8192), dim3(256), 0, stream>>>(kv_bf, Kt, Val);
        k_pack_r<<<dim3(2048), dim3(256), 0, stream>>>(r_f, Rt);

        // attention, chunked over batch b (heads batched inside)
        for (int b = 0; b < BSZ; ++b) {
            // content[h] = Qu[b,h] (CURxDH) @ Kt[b,h] (DHxFULL)
            gemm(Qu + (long)b * NH * CUR * DH, DH, (long)CUR * DH,
                 Kt + (long)b * NH * DH * FULLN, FULLN, (long)DH * FULLN,
                 cont_f, nullptr, FULLN, (long)CUR * FULLN,
                 nullptr, 0.f, CUR, FULLN, DH, NH, 0, 0);
            // position[h] = Qv[b,h] @ Rt[h]
            gemm(Qv + (long)b * NH * CUR * DH, DH, (long)CUR * DH,
                 Rt, FULLN, (long)DH * FULLN,
                 posr_f, nullptr, FULLN, (long)CUR * FULLN,
                 nullptr, 0.f, CUR, FULLN, DH, NH, 0, 0);
            // softmax((content + rel_shift(position)) * scale, mask)
            k_softmax<<<dim3(CUR, NH), dim3(256), 0, stream>>>(cont_f, posr_f, probs);
            // av[:, b, h*DH:] = probs[h] @ Val[b,h]
            gemm(probs, FULLN, (long)CUR * FULLN,
                 Val + (long)b * NH * FULLN * DH, DH, (long)FULLN * DH,
                 nullptr, av_bf + (long)b * NH * DH, (long)BSZ * NH * DH, DH,
                 nullptr, 0.f, CUR, DH, FULLN, NH, 0, 0);
        }

        // a1 = relu(av @ W_proj + b_proj)
        gemm(av_bf, HD, 0, Wpr_l, D, 0, nullptr, a1_bf, D, 0,
             b_proj + l * D, 1.0f, (int)Mc, (int)D, (int)HD, 1, 0, 1);

        // ---- GRU gate 1: xx = h, yy = a1 ----
        gemm(a1_bf, D, 0, G1(0), D, 0, rbuf, nullptr, D, 0, nullptr, 0.f, (int)Mc, (int)D, (int)D, 1, 0, 0);
        gemm(h_bf,  D, 0, G1(1), D, 0, rbuf, nullptr, D, 0, nullptr, 0.f, (int)Mc, (int)D, (int)D, 1, 1, 2);
        gemm(a1_bf, D, 0, G1(2), D, 0, zbuf, nullptr, D, 0, nullptr, 0.f, (int)Mc, (int)D, (int)D, 1, 0, 0);
        gemm(h_bf,  D, 0, G1(3), D, 0, zbuf, nullptr, D, 0, g1_b + l * D, -1.0f, (int)Mc, (int)D, (int)D, 1, 1, 2);
        gemm(a1_bf, D, 0, G1(4), D, 0, gbuf, nullptr, D, 0, nullptr, 0.f, (int)Mc, (int)D, (int)D, 1, 0, 0);
        k_ew_rx<<<dim3(ewBlocks), dim3(256), 0, stream>>>(rbuf, h_f, rx_bf, nEl);
        gemm(rx_bf, D, 0, G1(5), D, 0, gbuf, nullptr, D, 0, nullptr, 0.f, (int)Mc, (int)D, (int)D, 1, 1, 3);
        k_ew_gateout<<<dim3(ewBlocks), dim3(256), 0, stream>>>(zbuf, gbuf, h_f, o1_f, o1_bf, nEl);

        // x2 = LN(o1); m = relu(relu(x2@W_m1+b)@W_m2+b)
        k_ln<<<dim3((unsigned)Mc), dim3(256), 0, stream>>>(
            o1_f, (int)Mc, o1_f, ln2_g + l * D, ln2_b + l * D, x2_bf);
        gemm(x2_bf, D, 0, Wm1_l, D, 0, nullptr, m1_bf, D, 0, b_m1 + l * D, 1.0f, (int)Mc, (int)D, (int)D, 1, 0, 1);
        gemm(m1_bf, D, 0, Wm2_l, D, 0, nullptr, m_bf,  D, 0, b_m2 + l * D, 1.0f, (int)Mc, (int)D, (int)D, 1, 0, 1);

        // ---- GRU gate 2: xx = o1, yy = m ----
        gemm(m_bf,  D, 0, G2(0), D, 0, rbuf, nullptr, D, 0, nullptr, 0.f, (int)Mc, (int)D, (int)D, 1, 0, 0);
        gemm(o1_bf, D, 0, G2(1), D, 0, rbuf, nullptr, D, 0, nullptr, 0.f, (int)Mc, (int)D, (int)D, 1, 1, 2);
        gemm(m_bf,  D, 0, G2(2), D, 0, zbuf, nullptr, D, 0, nullptr, 0.f, (int)Mc, (int)D, (int)D, 1, 0, 0);
        gemm(o1_bf, D, 0, G2(3), D, 0, zbuf, nullptr, D, 0, g2_b + l * D, -1.0f, (int)Mc, (int)D, (int)D, 1, 1, 2);
        gemm(m_bf,  D, 0, G2(4), D, 0, gbuf, nullptr, D, 0, nullptr, 0.f, (int)Mc, (int)D, (int)D, 1, 0, 0);
        k_ew_rx<<<dim3(ewBlocks), dim3(256), 0, stream>>>(rbuf, o1_f, rx_bf, nEl);
        gemm(rx_bf, D, 0, G2(5), D, 0, gbuf, nullptr, D, 0, nullptr, 0.f, (int)Mc, (int)D, (int)D, 1, 1, 3);

        float* outf = (l == NL - 1) ? (float*)d_out : h_f;
        k_ew_gateout<<<dim3(ewBlocks), dim3(256), 0, stream>>>(zbuf, gbuf, o1_f, outf, h_bf, nEl);
    }
}
